// YOLOVHead_37898791419966
// MI455X (gfx1250) — compile-verified, matrix operand order checked by reading, not run on hardware
//
#include <hip/hip_runtime.h>
#include <hip/hip_bf16.h>

typedef unsigned short u16;
typedef __attribute__((ext_vector_type(16))) __bf16 v16bf;
typedef __attribute__((ext_vector_type(8)))  float  v8f;

// 16-bit A/B fragment (ISA 7.12.2): per lane, two contiguous 16B runs:
//   k = {0,2,4,6}+8*half (+1 in high bf16)  -> bytes [16*half, 16*half+16)
//   k = {16,18,20,22}+8*half                -> bytes [32+16*half, ...)
union Frag { v16bf v; uint4 q2[2]; unsigned int u[8]; u16 h[16]; };
union H8   { uint4 q; u16 h[8]; };

__device__ __forceinline__ u16 f2bf(float f) {
    unsigned int x = __float_as_uint(f);
    unsigned int r = x + 0x7FFFu + ((x >> 16) & 1u);
    return (u16)(r >> 16);
}
__device__ __forceinline__ float b2f(u16 h) {
    return __uint_as_float(((unsigned int)h) << 16);
}
__device__ __forceinline__ float dsilu(float f) { return f / (1.f + __expf(-f)); }

// ---- async global->LDS 16B copy (CDNA5 ASYNCcnt path), gated w/ fallback ----
#if __has_builtin(__builtin_amdgcn_global_load_async_to_lds_b128)
#define HAVE_ASYNC_LDS 1
typedef int v4i_g __attribute__((vector_size(4 * sizeof(int))));
typedef __attribute__((address_space(1))) v4i_g as1_v4i;
typedef __attribute__((address_space(3))) v4i_g as3_v4i;
#else
#define HAVE_ASYNC_LDS 0
#endif

__device__ __forceinline__ void copy16_g2l(const u16* g, u16* l) {
#if HAVE_ASYNC_LDS
    __builtin_amdgcn_global_load_async_to_lds_b128(
        (as1_v4i*)(unsigned long long)g,
        (as3_v4i*)(unsigned int)(unsigned long long)l, 0, 0);
#else
    *(uint4*)l = *(const uint4*)g;
#endif
}
__device__ __forceinline__ void copy_wait() {
#if HAVE_ASYNC_LDS
#if __has_builtin(__builtin_amdgcn_s_wait_asynccnt)
    __builtin_amdgcn_s_wait_asynccnt(0);
#else
    asm volatile("s_wait_asynccnt 0" ::: "memory");
#endif
#endif
}

// ---------------------------------------------------------------------------
// weight converters
// ---------------------------------------------------------------------------
__global__ void cvt_bf16(const float* __restrict__ s, u16* __restrict__ d, int n) {
    int i = blockIdx.x * 256 + threadIdx.x;
    if (i < n) d[i] = f2bf(s[i]);
}
// conv 3x3 weights: [co][ci][3][3] f32 -> [co][ tap*128 + ci ] bf16 (tap=ky*3+kx)
__global__ void cvt_conv_w(const float* __restrict__ s, u16* __restrict__ d) {
    int i = blockIdx.x * 256 + threadIdx.x;
    if (i < 128 * 1152) {
        int co = i / 1152, k = i - co * 1152;
        int r = k >> 7, ci = k & 127;
        d[i] = f2bf(s[co * 1152 + ci * 9 + r]);
    }
}
// transpose-convert: src [rows=K][cols=N] f32 -> dst [N][K] bf16
__global__ void cvt_bf16_t(const float* __restrict__ s, u16* __restrict__ d,
                           int rows, int cols) {
    int i = blockIdx.x * 256 + threadIdx.x;
    if (i < rows * cols) {
        int r = i / cols, c = i - r * cols;
        d[(size_t)c * rows + r] = f2bf(s[i]);
    }
}
// transpose + pad cols: src [rows=K][cols] -> dst [colsp][rows] bf16
__global__ void cvt_pad_t(const float* __restrict__ s, u16* __restrict__ d,
                          int rows, int cols, int colsp) {
    int i = blockIdx.x * 256 + threadIdx.x;
    if (i < rows * colsp) {
        int c = i / rows, r = i - c * rows;
        d[i] = (c < cols) ? f2bf(s[(size_t)r * cols + c]) : (u16)0;
    }
}

// ---------------------------------------------------------------------------
// Stem 1x1 conv (CI -> 128) + BN + SiLU, bf16 WMMA implicit GEMM.
// Input NCHW f32 (given); output NHWC bf16 [frame*HW + pos][128].
// ---------------------------------------------------------------------------
template <int NT>
__global__ __launch_bounds__(256)
void stem1x1_bn_silu(const float* __restrict__ in, const u16* __restrict__ wgt,
                     const float* __restrict__ scale, const float* __restrict__ bias,
                     u16* __restrict__ out, int HW, int CI) {
    extern __shared__ u16 spatch[];              // [NT*16][CI+8]
    const int KP = CI + 8;
    constexpr int POS = 16 * NT;
    const int pos0 = blockIdx.x * POS;
    const int frame = pos0 / HW;
    const int base = pos0 - frame * HW;

    // ci-major loop: consecutive lanes -> consecutive pixels (coalesced)
    for (int i = threadIdx.x; i < POS * CI; i += 256) {
        int ci = i / POS;
        int n = i - ci * POS;
        spatch[n * KP + ci] = f2bf(in[((size_t)(frame * CI + ci)) * HW + base + n]);
    }
    __syncthreads();

    const int wave = threadIdx.x >> 5;
    const int lane = threadIdx.x & 31;
    const int m = lane & 15;
    const int half = lane >> 4;
    const u16* wrow = wgt + (size_t)(wave * 16 + m) * CI;

    v8f acc[NT] = {};
    for (int kk = 0; kk < CI; kk += 32) {
        Frag a;
        a.q2[0] = *(const uint4*)(wrow + kk + 8 * half);
        a.q2[1] = *(const uint4*)(wrow + kk + 16 + 8 * half);
#pragma unroll
        for (int t = 0; t < NT; ++t) {
            Frag b;
            const u16* prow = spatch + (t * 16 + m) * KP + kk;
            b.q2[0] = *(const uint4*)(prow + 8 * half);
            b.q2[1] = *(const uint4*)(prow + 16 + 8 * half);
            acc[t] = __builtin_amdgcn_wmma_f32_16x16x32_bf16(
                false, a.v, false, b.v, (short)0, acc[t], false, false);
        }
    }
#pragma unroll
    for (int t = 0; t < NT; ++t) {
        int pos = base + t * 16 + (lane & 15);
        H8 o8;
#pragma unroll
        for (int vv = 0; vv < 8; ++vv) {
            int co = wave * 16 + (half << 3) + vv;
            float f = acc[t][vv] * scale[co] + bias[co];
            o8.h[vv] = f2bf(dsilu(f));
        }
        *(uint4*)(out + ((size_t)(frame * HW + pos)) * 128 + wave * 16 + (half << 3)) = o8.q;
    }
}

// ---------------------------------------------------------------------------
// 3x3 conv (128 -> 128, SAME) + BN + SiLU. NHWC bf16 in/out.
// K = 1152 with k = tap*128 + ci: each tap of each position is a contiguous
// 256B channel vector -> im2col fill is POS*9*16 aligned 16B copies, issued as
// global_load_async_to_lds_b128 (ASYNCcnt) when available.
// ---------------------------------------------------------------------------
template <int NT>
__global__ __launch_bounds__(256)
void conv3x3_bn_silu(const u16* __restrict__ in, const u16* __restrict__ wgt,
                     const float* __restrict__ scale, const float* __restrict__ bias,
                     u16* __restrict__ out, int H, int W) {
    constexpr int K = 1152, KP = 1160;
    constexpr int POS = 16 * NT;
    extern __shared__ u16 patch[];               // [POS][KP]
    const int HW = H * W;
    const int pos0 = blockIdx.x * POS;
    const int frame = pos0 / HW;
    const int base = pos0 - frame * HW;

    for (int i = threadIdx.x; i < POS * 9 * 16; i += 256) {
        int seg = i >> 4;                        // (n, tap)
        int chunk = i & 15;                      // 16B chunk of 256B vector
        int n = seg / 9;
        int r = seg - n * 9;
        int ky = r / 3, kx = r - ky * 3;
        int pos = base + n;
        int y = pos / W, x = pos - y * W;
        int yy = y + ky - 1, xx = x + kx - 1;
        u16* dst = patch + n * KP + r * 128 + chunk * 8;
        if (yy >= 0 && yy < H && xx >= 0 && xx < W) {
            const u16* src = in + ((size_t)(frame * HW + yy * W + xx)) * 128 + chunk * 8;
            copy16_g2l(src, dst);
        } else {
            *(uint4*)dst = make_uint4(0u, 0u, 0u, 0u);
        }
    }
    copy_wait();
    __syncthreads();

    const int wave = threadIdx.x >> 5;
    const int lane = threadIdx.x & 31;
    const int m = lane & 15;
    const int half = lane >> 4;
    const u16* wrow = wgt + (size_t)(wave * 16 + m) * K;

    v8f acc[NT] = {};
    for (int kk = 0; kk < K; kk += 32) {
        Frag a;
        a.q2[0] = *(const uint4*)(wrow + kk + 8 * half);
        a.q2[1] = *(const uint4*)(wrow + kk + 16 + 8 * half);
        __builtin_prefetch(wrow + kk + 32, 0, 3);
#pragma unroll
        for (int t = 0; t < NT; ++t) {
            Frag b;
            const u16* prow = patch + (t * 16 + m) * KP + kk;
            b.q2[0] = *(const uint4*)(prow + 8 * half);
            b.q2[1] = *(const uint4*)(prow + 16 + 8 * half);
            acc[t] = __builtin_amdgcn_wmma_f32_16x16x32_bf16(
                false, a.v, false, b.v, (short)0, acc[t], false, false);
        }
    }
#pragma unroll
    for (int t = 0; t < NT; ++t) {
        int pos = base + t * 16 + (lane & 15);
        H8 o8;
#pragma unroll
        for (int vv = 0; vv < 8; ++vv) {
            int co = wave * 16 + (half << 3) + vv;
            float f = acc[t][vv] * scale[co] + bias[co];
            o8.h[vv] = f2bf(dsilu(f));
        }
        *(uint4*)(out + ((size_t)(frame * HW + pos)) * 128 + wave * 16 + (half << 3)) = o8.q;
    }
}

// ---------------------------------------------------------------------------
// Generic bf16 WMMA GEMM, pre-transposed B: C[M,N] = act(A[M,K]*B + bias).
// ---------------------------------------------------------------------------
__global__ __launch_bounds__(32)
void gemm_bf16_wmma(const u16* __restrict__ A, const u16* __restrict__ Bt,
                    const float* __restrict__ bias, float* __restrict__ Cf,
                    u16* __restrict__ Cb, int N, int K, int act) {
    const int lane = threadIdx.x & 31;
    const int rr = lane & 15;
    const int half = lane >> 4;
    const int mt = blockIdx.x, nt = blockIdx.y;
    const u16* arow = A + (size_t)(mt * 16 + rr) * K;
    const u16* brow = Bt + (size_t)(nt * 16 + rr) * K;

    v8f acc = {};
    for (int kk = 0; kk < K; kk += 32) {
        Frag a, b;
        a.q2[0] = *(const uint4*)(arow + kk + 8 * half);
        a.q2[1] = *(const uint4*)(arow + kk + 16 + 8 * half);
        b.q2[0] = *(const uint4*)(brow + kk + 8 * half);
        b.q2[1] = *(const uint4*)(brow + kk + 16 + 8 * half);
        acc = __builtin_amdgcn_wmma_f32_16x16x32_bf16(
            false, a.v, false, b.v, (short)0, acc, false, false);
    }
    const int c = nt * 16 + (lane & 15);
    const float bv = bias ? bias[c] : 0.f;
#pragma unroll
    for (int vv = 0; vv < 8; ++vv) {
        int row = mt * 16 + vv + (half << 3);
        float f = acc[vv] + bv;
        if (act == 1) f = dsilu(f);
        size_t idx = (size_t)row * N + c;
        if (Cf) Cf[idx] = f;
        if (Cb) Cb[idx] = f2bf(f);
    }
}

// ---------------------------------------------------------------------------
// Pred heads + YOLOX decode. NHWC features: channels contiguous per anchor.
// ---------------------------------------------------------------------------
struct DecodeArgs {
    const u16* clsf;
    const u16* regf;
    const float* clspw[3]; const float* clspb[3];
    const float* regpw[3]; const float* regpb[3];
    const float* objpw[3]; const float* objpb[3];
    float* out;
    float* scores; float* clsmax; float* objs;
};

__global__ __launch_bounds__(256)
void decode_kernel(DecodeArgs A) {
    int g = blockIdx.x * 256 + threadIdx.x;
    if (g >= 16 * 3024) return;
    int f = g / 3024, a = g - f * 3024;
    int l, p, HW, Wd, str;
    if (a < 2304)      { l = 0; p = a;        HW = 2304; Wd = 48; str = 8;  }
    else if (a < 2880) { l = 1; p = a - 2304; HW = 576;  Wd = 24; str = 16; }
    else               { l = 2; p = a - 2880; HW = 144;  Wd = 12; str = 32; }
    size_t lo = (l == 0) ? 0ull : ((l == 1) ? 4718592ull : 5898240ull);
    const u16* cfp = A.clsf + lo + ((size_t)(f * HW + p)) * 128;
    const u16* rfp = A.regf + lo + ((size_t)(f * HW + p)) * 128;

    const float* cw = A.clspw[l]; const float* cb = A.clspb[l];
    float acc[30];
#pragma unroll
    for (int u = 0; u < 30; ++u) acc[u] = cb[u];
    for (int c = 0; c < 128; ++c) {
        float x = b2f(cfp[c]);
#pragma unroll
        for (int u = 0; u < 30; ++u) acc[u] += x * cw[u * 128 + c];
    }
    const float* rw = A.regpw[l]; const float* rb = A.regpb[l];
    const float* ow = A.objpw[l]; const float* ob = A.objpb[l];
    float racc[5] = {rb[0], rb[1], rb[2], rb[3], ob[0]};
    for (int c = 0; c < 128; ++c) {
        float x = b2f(rfp[c]);
        racc[0] += x * rw[c];       racc[1] += x * rw[128 + c];
        racc[2] += x * rw[256 + c]; racc[3] += x * rw[384 + c];
        racc[4] += x * ow[c];
    }
    int y = p / Wd, x_ = p - y * Wd;
    float* o = A.out + (size_t)g * 35;
    o[0] = (racc[0] + (float)x_) * (float)str;
    o[1] = (racc[1] + (float)y) * (float)str;
    o[2] = __expf(racc[2]) * (float)str;
    o[3] = __expf(racc[3]) * (float)str;
    float ov = 1.f / (1.f + __expf(-racc[4]));
    o[4] = ov;
    float cmax = 0.f;
#pragma unroll
    for (int u = 0; u < 30; ++u) {
        float s = 1.f / (1.f + __expf(-acc[u]));
        o[5 + u] = s;
        if (s > cmax) cmax = s;
    }
    A.scores[g] = ov * cmax; A.clsmax[g] = cmax; A.objs[g] = ov;
}

// ---------------------------------------------------------------------------
// Per-frame top-30 selection + NHWC feature gather.
// ---------------------------------------------------------------------------
__global__ __launch_bounds__(256)
void topk_gather(const float* __restrict__ scores, const float* __restrict__ clsmax,
                 const float* __restrict__ objs, const u16* __restrict__ cls2f,
                 float* __restrict__ xf, float* __restrict__ sw) {
    __shared__ float sc[3024];
    __shared__ float sv[256];
    __shared__ int   si[256];
    __shared__ int   chosen;
    const int f = blockIdx.x, t = threadIdx.x;
    for (int a = t; a < 3024; a += 256) sc[a] = scores[f * 3024 + a];
    __syncthreads();
    for (int r = 0; r < 30; ++r) {
        float best = -3.0e38f; int bi = 0;
        for (int a = t; a < 3024; a += 256) {
            float s = sc[a];
            if (s > best) { best = s; bi = a; }
        }
        sv[t] = best; si[t] = bi;
        __syncthreads();
        for (int o = 128; o > 0; o >>= 1) {
            if (t < o && sv[t + o] > sv[t]) { sv[t] = sv[t + o]; si[t] = si[t + o]; }
            __syncthreads();
        }
        if (t == 0) {
            chosen = si[0];
            sc[si[0]] = -3.0e38f;
            int g = f * 3024 + si[0];
            sw[f * 30 + r] = clsmax[g] * objs[g];
        }
        __syncthreads();
        int a = chosen;
        int l, p, HW;
        if (a < 2304)      { l = 0; p = a;        HW = 2304; }
        else if (a < 2880) { l = 1; p = a - 2304; HW = 576;  }
        else               { l = 2; p = a - 2880; HW = 144;  }
        size_t lo = (l == 0) ? 0ull : ((l == 1) ? 4718592ull : 5898240ull);
        if (t < 128)
            xf[((size_t)(f * 30 + r)) * 128 + t] =
                b2f(cls2f[lo + ((size_t)(f * HW + p)) * 128 + t]);
        __syncthreads();
    }
}

// Row L2 normalization (cosine gate) + bf16 copy of xf.
__global__ __launch_bounds__(128)
void rownorm_cvt(const float* __restrict__ xf, float* __restrict__ fn,
                 u16* __restrict__ xb) {
    __shared__ float s[128];
    int r = blockIdx.x, t = threadIdx.x;
    float x = xf[(size_t)r * 128 + t];
    s[t] = x * x;
    __syncthreads();
    for (int o = 64; o > 0; o >>= 1) {
        if (t < o) s[t] += s[t + o];
        __syncthreads();
    }
    float inv = 1.f / (sqrtf(s[0]) + 1e-6f);
    fn[(size_t)r * 128 + t] = x * inv;
    xb[(size_t)r * 128 + t] = f2bf(x);
}

// ---------------------------------------------------------------------------
// MSA: per (head, query-row) masked/score-weighted softmax + aggregation.
// ---------------------------------------------------------------------------
__global__ __launch_bounds__(256)
void attn_kernel(const float* __restrict__ q, const float* __restrict__ k,
                 const float* __restrict__ v, const float* __restrict__ fn,
                 const float* __restrict__ sw, float* __restrict__ agg) {
    __shared__ float rowbuf[8][480];
    const int wave = threadIdx.x >> 5;
    const int lane = threadIdx.x & 31;
    const int rid = blockIdx.x * 8 + wave;
    const int h = rid / 480;
    const int n = rid - h * 480;
    const float scl = 0.17677669529663687f;      // 32^-0.5
    float* row = rowbuf[wave];

    for (int mb = 0; mb < 15; ++mb) {
        int m = mb * 32 + lane;
        float s = 0.f, c = 0.f;
        for (int d = 0; d < 32; ++d)
            s += q[(size_t)n * 128 + h * 32 + d] * k[(size_t)m * 128 + h * 32 + d];
        for (int d = 0; d < 128; ++d)
            c += fn[(size_t)n * 128 + d] * fn[(size_t)m * 128 + d];
        row[m] = (c >= 0.75f) ? s * scl : -1.0e4f;
    }
    float mx = -3.0e38f;
    for (int mb = 0; mb < 15; ++mb) mx = fmaxf(mx, row[mb * 32 + lane]);
    for (int o = 16; o > 0; o >>= 1) mx = fmaxf(mx, __shfl_xor(mx, o, 32));
    float s1 = 0.f;
    for (int mb = 0; mb < 15; ++mb) {
        int m = mb * 32 + lane;
        float e = __expf(row[m] - mx);
        row[m] = e; s1 += e;
    }
    for (int o = 16; o > 0; o >>= 1) s1 += __shfl_xor(s1, o, 32);
    float inv1 = 1.f / s1;
    float s2 = 0.f;
    for (int mb = 0; mb < 15; ++mb) {
        int m = mb * 32 + lane;
        float a = row[m] * inv1 * sw[m];
        row[m] = a; s2 += a;
    }
    for (int o = 16; o > 0; o >>= 1) s2 += __shfl_xor(s2, o, 32);
    float inv2 = 1.f / (s2 + 1e-6f);
    float acc = 0.f;
    for (int m = 0; m < 480; ++m)
        acc += row[m] * v[(size_t)m * 128 + h * 32 + lane];
    agg[(size_t)n * 128 + h * 32 + lane] = acc * inv2;
}

// concat(xf, agg) -> bf16 [480][256]
__global__ void concat_cat(const u16* __restrict__ xb, const float* __restrict__ agg,
                           u16* __restrict__ cat) {
    int i = blockIdx.x * 256 + threadIdx.x;
    if (i >= 480 * 256) return;
    int r = i >> 8, c = i & 255;
    cat[i] = (c < 128) ? xb[r * 128 + c] : f2bf(agg[r * 128 + (c - 128)]);
}

// copy [480][32]-padded logits into d_out [480][30], add bias
__global__ void logits_copy(const float* __restrict__ src, const float* __restrict__ bias,
                            float* __restrict__ dst) {
    int i = blockIdx.x * 256 + threadIdx.x;
    if (i >= 480 * 30) return;
    int r = i / 30, c = i - r * 30;
    dst[i] = src[r * 32 + c] + bias[c];
}

// ===========================================================================
extern "C" void kernel_launch(void* const* d_in, const int* in_sizes, int n_in,
                              void* d_out, int out_size, void* d_ws, size_t ws_size,
                              hipStream_t stream) {
    (void)in_sizes; (void)n_in; (void)out_size; (void)ws_size;
    const int Bn = 16;
    const int CIs[3] = {128, 256, 512};
    const int Hs[3]  = {48, 24, 12};
    const int Ps[3]  = {2304, 576, 144};
    const size_t loff[3] = {0ull, 4718592ull, 5898240ull};   // bf16 elem offsets

    auto P = [&](int i) -> const float* { return (const float*)d_in[i]; };
    auto pidx = [&](int l, int o) -> int { return 3 + 27 * l + o; };

    size_t off = 0;
    auto alloc = [&](size_t bytes) -> char* {
        char* p = (char*)d_ws + off;
        off += (bytes + 255) & ~(size_t)255;
        return p;
    };

    // ---- workspace ----
    u16* wstem[3];
    for (int l = 0; l < 3; ++l) wstem[l] = (u16*)alloc((size_t)128 * CIs[l] * 2);
    u16* wconv[18];
    for (int i = 0; i < 18; ++i) wconv[i] = (u16*)alloc((size_t)147456 * 2);
    u16* wq = (u16*)alloc(16384 * 2);
    u16* wk = (u16*)alloc(16384 * 2);
    u16* wv = (u16*)alloc(16384 * 2);
    u16* wl1 = (u16*)alloc(65536 * 2);
    u16* wl2 = (u16*)alloc(131072 * 2);
    u16* wcp = (u16*)alloc((size_t)512 * 32 * 2);

    const size_t ACT = (size_t)128 * Bn * 3024;
    u16* s_bf  = (u16*)alloc(ACT * 2);
    u16* tmp_b = (u16*)alloc(ACT * 2);
    u16* clsf  = (u16*)alloc(ACT * 2);
    u16* regf  = (u16*)alloc(ACT * 2);
    u16* cls2f = (u16*)alloc(ACT * 2);

    float* scores = (float*)alloc((size_t)48384 * 4);
    float* clsmax = (float*)alloc((size_t)48384 * 4);
    float* objs   = (float*)alloc((size_t)48384 * 4);
    float* xf  = (float*)alloc((size_t)480 * 128 * 4);
    float* fn  = (float*)alloc((size_t)480 * 128 * 4);
    float* swv = (float*)alloc((size_t)480 * 4);
    u16*   xb  = (u16*)alloc((size_t)480 * 128 * 2);
    float* qf  = (float*)alloc((size_t)480 * 128 * 4);
    float* kf  = (float*)alloc((size_t)480 * 128 * 4);
    float* vf  = (float*)alloc((size_t)480 * 128 * 4);
    float* agg = (float*)alloc((size_t)480 * 128 * 4);
    u16*   cat = (u16*)alloc((size_t)480 * 256 * 2);
    u16*   l1o = (u16*)alloc((size_t)480 * 256 * 2);
    u16*   l2o = (u16*)alloc((size_t)480 * 512 * 2);
    float* lg32 = (float*)alloc((size_t)480 * 32 * 4);

    // ---- weight conversions ----
    for (int l = 0; l < 3; ++l) {
        int n = 128 * CIs[l];
        cvt_bf16<<<(n + 255) / 256, 256, 0, stream>>>(P(pidx(l, 0)), wstem[l], n);
        for (int bi = 0; bi < 6; ++bi)
            cvt_conv_w<<<(147456 + 255) / 256, 256, 0, stream>>>(
                P(pidx(l, 3 + 3 * bi)), wconv[l * 6 + bi]);
    }
    cvt_bf16_t<<<64, 256, 0, stream>>>(P(84), wq, 128, 128);
    cvt_bf16_t<<<64, 256, 0, stream>>>(P(85), wk, 128, 128);
    cvt_bf16_t<<<64, 256, 0, stream>>>(P(86), wv, 128, 128);
    cvt_bf16_t<<<256, 256, 0, stream>>>(P(87), wl1, 256, 256);
    cvt_bf16_t<<<512, 256, 0, stream>>>(P(89), wl2, 256, 512);
    cvt_pad_t<<<(512 * 32 + 255) / 256, 256, 0, stream>>>(P(91), wcp, 512, 30, 32);

    // ---- stems + conv towers ----
    for (int l = 0; l < 3; ++l) {
        const int NT = (l == 2) ? 1 : 2;
        const int POS = 16 * NT;
        int tiles = Bn * Ps[l] / POS;
        size_t shs = (size_t)POS * (CIs[l] + 8) * 2;
        size_t shc = (size_t)POS * 1160 * 2;

        if (NT == 2)
            stem1x1_bn_silu<2><<<tiles, 256, shs, stream>>>(
                P(l), wstem[l], P(pidx(l, 1)), P(pidx(l, 2)), s_bf + loff[l], Ps[l], CIs[l]);
        else
            stem1x1_bn_silu<1><<<tiles, 256, shs, stream>>>(
                P(l), wstem[l], P(pidx(l, 1)), P(pidx(l, 2)), s_bf + loff[l], Ps[l], CIs[l]);

        const u16* w0 = wconv[l * 6 + 0]; const u16* w1 = wconv[l * 6 + 1];
        const u16* w2 = wconv[l * 6 + 2]; const u16* w3 = wconv[l * 6 + 3];
        const u16* w4 = wconv[l * 6 + 4]; const u16* w5 = wconv[l * 6 + 5];
        if (NT == 2) {
            conv3x3_bn_silu<2><<<tiles, 256, shc, stream>>>(s_bf + loff[l], w0,
                P(pidx(l, 4)), P(pidx(l, 5)), tmp_b + loff[l], Hs[l], Hs[l]);
            conv3x3_bn_silu<2><<<tiles, 256, shc, stream>>>(tmp_b + loff[l], w1,
                P(pidx(l, 7)), P(pidx(l, 8)), clsf + loff[l], Hs[l], Hs[l]);
            conv3x3_bn_silu<2><<<tiles, 256, shc, stream>>>(s_bf + loff[l], w2,
                P(pidx(l, 10)), P(pidx(l, 11)), tmp_b + loff[l], Hs[l], Hs[l]);
            conv3x3_bn_silu<2><<<tiles, 256, shc, stream>>>(tmp_b + loff[l], w3,
                P(pidx(l, 13)), P(pidx(l, 14)), regf + loff[l], Hs[l], Hs[l]);
            conv3x3_bn_silu<2><<<tiles, 256, shc, stream>>>(s_bf + loff[l], w4,
                P(pidx(l, 16)), P(pidx(l, 17)), tmp_b + loff[l], Hs[l], Hs[l]);
            conv3x3_bn_silu<2><<<tiles, 256, shc, stream>>>(tmp_b + loff[l], w5,
                P(pidx(l, 19)), P(pidx(l, 20)), cls2f + loff[l], Hs[l], Hs[l]);
        } else {
            conv3x3_bn_silu<1><<<tiles, 256, shc, stream>>>(s_bf + loff[l], w0,
                P(pidx(l, 4)), P(pidx(l, 5)), tmp_b + loff[l], Hs[l], Hs[l]);
            conv3x3_bn_silu<1><<<tiles, 256, shc, stream>>>(tmp_b + loff[l], w1,
                P(pidx(l, 7)), P(pidx(l, 8)), clsf + loff[l], Hs[l], Hs[l]);
            conv3x3_bn_silu<1><<<tiles, 256, shc, stream>>>(s_bf + loff[l], w2,
                P(pidx(l, 10)), P(pidx(l, 11)), tmp_b + loff[l], Hs[l], Hs[l]);
            conv3x3_bn_silu<1><<<tiles, 256, shc, stream>>>(tmp_b + loff[l], w3,
                P(pidx(l, 13)), P(pidx(l, 14)), regf + loff[l], Hs[l], Hs[l]);
            conv3x3_bn_silu<1><<<tiles, 256, shc, stream>>>(s_bf + loff[l], w4,
                P(pidx(l, 16)), P(pidx(l, 17)), tmp_b + loff[l], Hs[l], Hs[l]);
            conv3x3_bn_silu<1><<<tiles, 256, shc, stream>>>(tmp_b + loff[l], w5,
                P(pidx(l, 19)), P(pidx(l, 20)), cls2f + loff[l], Hs[l], Hs[l]);
        }
    }

    // ---- pred heads + decode ----
    DecodeArgs da;
    da.clsf = clsf; da.regf = regf;
    for (int l = 0; l < 3; ++l) {
        da.clspw[l] = P(pidx(l, 21)); da.clspb[l] = P(pidx(l, 22));
        da.regpw[l] = P(pidx(l, 23)); da.regpb[l] = P(pidx(l, 24));
        da.objpw[l] = P(pidx(l, 25)); da.objpb[l] = P(pidx(l, 26));
    }
    da.out = (float*)d_out; da.scores = scores; da.clsmax = clsmax; da.objs = objs;
    decode_kernel<<<189, 256, 0, stream>>>(da);

    // ---- top-K + MSA + MLP ----
    topk_gather<<<16, 256, 0, stream>>>(scores, clsmax, objs, cls2f, xf, swv);
    rownorm_cvt<<<480, 128, 0, stream>>>(xf, fn, xb);

    gemm_bf16_wmma<<<dim3(30, 8), 32, 0, stream>>>(xb, wq, nullptr, qf, nullptr, 128, 128, 0);
    gemm_bf16_wmma<<<dim3(30, 8), 32, 0, stream>>>(xb, wk, nullptr, kf, nullptr, 128, 128, 0);
    gemm_bf16_wmma<<<dim3(30, 8), 32, 0, stream>>>(xb, wv, nullptr, vf, nullptr, 128, 128, 0);

    attn_kernel<<<240, 256, 0, stream>>>(qf, kf, vf, fn, swv, agg);
    concat_cat<<<480, 256, 0, stream>>>(xb, agg, cat);

    gemm_bf16_wmma<<<dim3(30, 16), 32, 0, stream>>>(cat, wl1, P(88), nullptr, l1o, 256, 256, 1);
    gemm_bf16_wmma<<<dim3(30, 32), 32, 0, stream>>>(l1o, wl2, P(90), nullptr, l2o, 512, 256, 0);
    gemm_bf16_wmma<<<dim3(30, 2), 32, 0, stream>>>(l2o, wcp, nullptr, lg32, nullptr, 32, 512, 0);

    float* logits_out = (float*)d_out + (size_t)16 * 3024 * 35;
    logits_copy<<<(480 * 30 + 255) / 256, 256, 0, stream>>>(lg32, P(92), logits_out);
}
